// GraphSageConvSoftmax_65944927863276
// MI455X (gfx1250) — matrix operand order repaired and checked by reading, not compiled, not run on hardware
//
#include <hip/hip_runtime.h>

typedef __attribute__((ext_vector_type(16))) _Float16 v16h;
typedef __attribute__((ext_vector_type(8)))  _Float16 v8h;
typedef __attribute__((ext_vector_type(4)))  _Float16 v4h;
typedef __attribute__((ext_vector_type(8)))  float    v8f;
typedef __attribute__((ext_vector_type(4)))  unsigned int v4u;
typedef __attribute__((ext_vector_type(8)))  int      v8i;
typedef __attribute__((ext_vector_type(4)))  int      v4i;

// explicit LDS (addrspace 3) views of TDM-written buffers: keeps the loads on
// the ds_* path (32-bit addressing) while the laundered offset hides provenance
typedef __attribute__((address_space(3))) const float    lds_cf;
typedef __attribute__((address_space(3))) const v8h      lds_cv8h;

#define NNODES 4096
#define FDIM   64
#define CHOUT  128
#define BATCH  4
#define NCHUNK (NNODES / 32)      // 128 K-chunks of 32
#define FT_H   (FDIM * 40)        // one pre-transposed f16 feature tile: 64 rows x 40 halfs

// ---------------------------------------------------------------------------
// helpers
// ---------------------------------------------------------------------------
__device__ static inline unsigned lds_off(const void* p) {
  // low 32 bits of a generic shared pointer == LDS byte offset (plain, foldable)
  return (unsigned)(uintptr_t)p;
}

__device__ static inline unsigned lds_off_opaque(const void* p) {
  // same, but opaque so the optimizer can't fold loads from TDM-written arrays
  unsigned x = (unsigned)(uintptr_t)p;
  asm volatile("" : "+v"(x));
  return x;
}

__device__ static inline v16h frag16(const _Float16* p0, const _Float16* p1) {
  // two aligned 16B LDS loads -> one 16-half WMMA fragment
  v8h lo = *(const v8h*)p0;
  v8h hh = *(const v8h*)p1;
  return __builtin_shufflevector(lo, hh, 0,1,2,3,4,5,6,7,8,9,10,11,12,13,14,15);
}

__device__ static inline v16h frag16_lds(unsigned a0, unsigned a1) {
  v8h lo = *(lds_cv8h*)(uintptr_t)a0;   // ds_load_b128
  v8h hh = *(lds_cv8h*)(uintptr_t)a1;   // ds_load_b128
  return __builtin_shufflevector(lo, hh, 0,1,2,3,4,5,6,7,8,9,10,11,12,13,14,15);
}

__device__ static inline v8f wmma16(v16h a, v16h b, v8f c) {
  return __builtin_amdgcn_wmma_f32_16x16x32_f16(false, a, false, b,
                                                (short)0, c, false, false);
}

// Issue one TDM tile load (up to 2-D), descriptor packed per ISA §8.3/§8.4.
__device__ static inline void tdm_load(unsigned lds_byteoff, const void* gptr,
                                       unsigned data_size_code,
                                       unsigned tensor_d0, unsigned tensor_d1,
                                       unsigned tile_d0, unsigned tile_d1,
                                       unsigned long long stride0) {
  unsigned long long ga = (unsigned long long)(uintptr_t)gptr;
  v4u g0;
  g0.x = 1u;                                   // count=1, user descriptor
  g0.y = lds_byteoff;                          // lds_addr
  g0.z = (unsigned)(ga & 0xffffffffu);         // global_addr[31:0]
  g0.w = (unsigned)((ga >> 32) & 0x1ffffffu)   // global_addr[56:32]
       | (2u << 30);                           // type=2 (image)
  v8i g1;
  g1[0] = (int)(data_size_code << 16);                                   // mask=0, ds
  g1[1] = (int)((tensor_d0 & 0xffffu) << 16);                            // dim0[15:0]
  g1[2] = (int)(((tensor_d0 >> 16) & 0xffffu) | ((tensor_d1 & 0xffffu) << 16));
  g1[3] = (int)(((tensor_d1 >> 16) & 0xffffu) | ((tile_d0 & 0xffffu) << 16));
  g1[4] = (int)(tile_d1 & 0xffffu);                                      // tile_dim2=0
  g1[5] = (int)(stride0 & 0xffffffffu);
  g1[6] = (int)((stride0 >> 32) & 0xffffu);                              // stride1=0
  g1[7] = 0;
  v4i z = {0, 0, 0, 0};
#if __clang_major__ >= 23
  v8i z8 = {0, 0, 0, 0, 0, 0, 0, 0};
  __builtin_amdgcn_tensor_load_to_lds(g0, g1, z, z, z8, 0);
#else
  __builtin_amdgcn_tensor_load_to_lds(g0, g1, z, z, 0);
#endif
}

// ---------------------------------------------------------------------------
// Pre-pass: features f32 [B,N,64] -> f16 tiles [B][128][64][40] (N-major,
// row-padded to 80B) so phase-1 B-tiles are single contiguous TDM copies and
// fragments are ds_load_b128-able.
// ---------------------------------------------------------------------------
__global__ __launch_bounds__(128) void cvt_features(
    const float* __restrict__ feat, _Float16* __restrict__ featHT) {
  const int b = blockIdx.y, jc = blockIdx.x;
  const size_t srcBase = ((size_t)b * NNODES + (size_t)jc * 32) * FDIM;
  _Float16* dst = featHT + ((size_t)b * NCHUNK + jc) * FT_H;
  #pragma unroll
  for (int it = 0; it < 16; ++it) {
    int idx = threadIdx.x + it * 128;       // 0..2047
    int c = idx & 63, jj = idx >> 6;        // feature col, j-in-chunk
    dst[c * 40 + jj] = (_Float16)feat[srcBase + (size_t)jj * FDIM + c];
  }
}

// ---------------------------------------------------------------------------
// Phase 1: fused masked softmax over fltr^T + SpMM (w @ features).
// Grid (N/16, B), 128 threads. TDM double-buffers the fltr slab (32x16 f32,
// 2-D tile) and the pre-transposed feature tile (5120B, 1-D) into LDS;
// wave 0 issues/waits (s_wait_tensorcnt), barriers publish to all waves.
// ---------------------------------------------------------------------------
__global__ __launch_bounds__(128) void gsage_softmax_spmm(
    const float* __restrict__ fltr,        // [B,N,N]
    const _Float16* __restrict__ featHT,   // [B][128][64][40]
    float* __restrict__ neigh)             // [B,N,64]
{
  __shared__ float    ldsFltr[2][32][16];  // TDM dest: fltr slab (j-major)
  __shared__ _Float16 ldsFeat[2][FDIM][40];// TDM dest: B tile (N-major)
  __shared__ _Float16 expT[16][40];        // A tile: exp values [i][k]
  __shared__ float    sDen[16];

  const int b  = blockIdx.y;
  const int i0 = blockIdx.x * 16;
  const int tid = threadIdx.x, lane = tid & 31, wv = tid >> 5;
  const int n = lane & 15, hi = lane >> 4;
  const int ei = tid & 15;                 // exp thread's fixed i column
  const int ej = (tid >> 4) * 4;           // exp thread's j base (0..28)

  if (tid < 16) sDen[tid] = 0.0f;
  float dacc = 0.0f;
  v8f c = {};

  const float*    fltrBase = fltr + (size_t)b * NNODES * NNODES + i0;
  const _Float16* featBase = featHT + (size_t)b * NCHUNK * FT_H;

  // opaque read-bases for TDM-written buffers, hoisted out of the loop
  const unsigned fltrRd[2] = { lds_off_opaque(&ldsFltr[0][0][0]),
                               lds_off_opaque(&ldsFltr[1][0][0]) };
  const unsigned featRd[2] = { lds_off_opaque(&ldsFeat[0][0][0]),
                               lds_off_opaque(&ldsFeat[1][0][0]) };

  auto issue = [&](int jc) {
    // plain (foldable) LDS offsets: descriptor stays on the scalar path
    unsigned dA = lds_off(&ldsFltr[jc & 1][0][0]);
    unsigned dB = lds_off(&ldsFeat[jc & 1][0][0]);
    // 32x16 f32 tile out of the [N,N] tensor, row stride N
    tdm_load(dA, fltrBase + (size_t)jc * 32 * NNODES,
             /*4B*/2, NNODES, NNODES, /*tile*/16, 32, NNODES);
    // contiguous 5120B feature tile (treated as 640 x 8B, 1-D)
    tdm_load(dB, featBase + (size_t)jc * FT_H,
             /*8B*/3, 640, 1, 640, 1, 640);
  };
  if (wv == 0) { issue(0); issue(1); }

  for (int jc = 0; jc < NCHUNK; ++jc) {
    const int s = jc & 1;
    if (wv == 0) {
      if (jc < NCHUNK - 1) {
        __builtin_amdgcn_s_wait_tensorcnt(2);   // chunk jc complete, jc+1 in flight
      } else {
        __builtin_amdgcn_s_wait_tensorcnt(0);   // final chunk complete
      }
    }
    __syncthreads();                       // buffers[s] ready for all waves

    // build A tile: exp+mask of the transposed fltr slab, f32 -> f16
    {
      lds_cf* pf = (lds_cf*)(uintptr_t)fltrRd[s];
      v4h eh;
      #pragma unroll
      for (int x = 0; x < 4; ++x) {
        float v = pf[(ej + x) * 16 + ei];  // ds_load_b32 / ds_load_2addr_b32
        float e = (v != 0.0f) ? __expf(v) : 0.0f;
        eh[x] = (_Float16)e;
        dacc += e;
      }
      *(v4h*)&expT[ei][ej] = eh;           // one 8B LDS store
    }
    __syncthreads();                       // expT published

    // fragments: 2x ds_load_b128 each, then WMMA
    {
      const _Float16* pe = &expT[0][0];
      v16h a = frag16(pe + n * 40 + 16 * hi,
                      pe + n * 40 + 16 * hi + 8);
      unsigned bbase = featRd[s]
                     + (unsigned)(((wv * 16 + n) * 40 + 16 * hi) * 2);
      v16h bf = frag16_lds(bbase, bbase + 16);
      c = wmma16(a, bf, c);
    }
    __syncthreads();                       // all reads of buffers[s] done

    if (wv == 0 && jc + 2 < NCHUNK) issue(jc + 2);
  }

  atomicAdd(&sDen[ei], dacc);              // single ds_add_f32 flush
  __syncthreads();

  #pragma unroll
  for (int r = 0; r < 8; ++r) {            // C/D: M = 8*hi + r, N = lane&15
    int M = hi * 8 + r;
    neigh[((size_t)b * NNODES + i0 + M) * FDIM + wv * 16 + n] =
        c[r] * (1.0f / sDen[M]);
  }
}

// ---------------------------------------------------------------------------
// Phase 2: out = relu(concat(features, neigh) @ kernel + bias).
// kernel staged N-major in LDS (kT[c][k]) so B fragments are contiguous.
// ---------------------------------------------------------------------------
__global__ __launch_bounds__(128) void gsage_dense_out(
    const float* __restrict__ feat,    // [B,N,64]
    const float* __restrict__ neigh,   // [B,N,64]
    const float* __restrict__ kernelW, // [128,128]
    const float* __restrict__ bias,    // [128]
    float* __restrict__ out)           // [B,N,128]
{
  __shared__ _Float16 kT[CHOUT][136];  // [out-channel][k], rows 272B (16B mult)
  __shared__ _Float16 aT[16][136];     // [m][k]
  __shared__ float    sBias[CHOUT];

  const int b  = blockIdx.y;
  const int m0 = blockIdx.x * 16;
  const int tid = threadIdx.x, lane = tid & 31, wv = tid >> 5;
  const int n = lane & 15, hi = lane >> 4;

  // stage kernel, transposed to N-major: 4096 float4 / 128 threads
  #pragma unroll
  for (int it = 0; it < 32; ++it) {
    int idx = tid + it * 128;          // 0..4095
    int f  = idx >> 5;                 // k row 0..127
    int c4 = (idx & 31) * 4;           // out-channel base
    float4 v = *(const float4*)(kernelW + f * CHOUT + c4);
    kT[c4 + 0][f] = (_Float16)v.x;
    kT[c4 + 1][f] = (_Float16)v.y;
    kT[c4 + 2][f] = (_Float16)v.z;
    kT[c4 + 3][f] = (_Float16)v.w;
  }
  sBias[tid] = bias[tid];

  // stage A = concat(features, neigh) rows m0..m0+15
  #pragma unroll
  for (int it = 0; it < 4; ++it) {
    int idx = tid + it * 128;          // 0..511 float4s
    int mm = idx >> 5;                 // 0..15
    int c4 = (idx & 31) * 4;           // 0..124
    const float* src = (c4 < FDIM)
        ? (feat  + ((size_t)b * NNODES + m0 + mm) * FDIM + c4)
        : (neigh + ((size_t)b * NNODES + m0 + mm) * FDIM + (c4 - FDIM));
    float4 v = *(const float4*)src;
    v4h h; h[0] = (_Float16)v.x; h[1] = (_Float16)v.y;
           h[2] = (_Float16)v.z; h[3] = (_Float16)v.w;
    *(v4h*)&aT[mm][c4] = h;            // 8B-aligned store
  }
  __syncthreads();

  v8f c0 = {}, c1 = {};
  const int nt0 = wv * 2, nt1 = wv * 2 + 1;
  const _Float16* pa = &aT[0][0];
  const _Float16* pk = &kT[0][0];

  #pragma unroll
  for (int kc = 0; kc < 4; ++kc) {
    v16h a  = frag16(pa + n * 136 + kc * 32 + 16 * hi,
                     pa + n * 136 + kc * 32 + 16 * hi + 8);
    v16h b0 = frag16(pk + (nt0 * 16 + n) * 136 + kc * 32 + 16 * hi,
                     pk + (nt0 * 16 + n) * 136 + kc * 32 + 16 * hi + 8);
    v16h b1 = frag16(pk + (nt1 * 16 + n) * 136 + kc * 32 + 16 * hi,
                     pk + (nt1 * 16 + n) * 136 + kc * 32 + 16 * hi + 8);
    c0 = wmma16(a, b0, c0);
    c1 = wmma16(a, b1, c1);
  }

  #pragma unroll
  for (int r = 0; r < 8; ++r) {
    int M = hi * 8 + r;
    size_t rowBase = ((size_t)b * NNODES + m0 + M) * CHOUT;
    float v0 = c0[r] + sBias[nt0 * 16 + n];
    float v1 = c1[r] + sBias[nt1 * 16 + n];
    out[rowBase + nt0 * 16 + n] = v0 > 0.0f ? v0 : 0.0f;
    out[rowBase + nt1 * 16 + n] = v1 > 0.0f ? v1 : 0.0f;
  }
}

// ---------------------------------------------------------------------------
extern "C" void kernel_launch(void* const* d_in, const int* in_sizes, int n_in,
                              void* d_out, int out_size, void* d_ws, size_t ws_size,
                              hipStream_t stream) {
  const float* feat = (const float*)d_in[0];  // [B,N,F]
  const float* fltr = (const float*)d_in[1];  // [B,N,N]
  const float* kern = (const float*)d_in[2];  // [2F,CH]
  const float* bias = (const float*)d_in[3];  // [CH]
  float* out = (float*)d_out;                 // [B,N,CH]

  float*    neigh  = (float*)d_ws;                                 // 4 MiB
  _Float16* featHT = (_Float16*)((char*)d_ws + ((size_t)4 << 20)); // 2.5 MiB

  dim3 gridC(NCHUNK, BATCH);
  cvt_features<<<gridC, 128, 0, stream>>>(feat, featHT);
  dim3 grid(NNODES / 16, BATCH);
  gsage_softmax_spmm<<<grid, 128, 0, stream>>>(fltr, featHT, neigh);
  gsage_dense_out<<<grid, 128, 0, stream>>>(feat, neigh, kern, bias, out);
}
